// HyperLinearV1_72078141161613
// MI455X (gfx1250) — matrix operand order repaired and checked by reading, not compiled, not run on hardware
//
#include <hip/hip_runtime.h>
#include <hip/hip_bf16.h>

typedef __attribute__((ext_vector_type(16))) _Float16 v16h;
typedef __attribute__((ext_vector_type(4)))  _Float16 v4h;
typedef __attribute__((ext_vector_type(8)))  float    v8f;

#define DIN     8192
#define DOUT    8192
#define K1      4096      // DIN/2 index columns
#define NTOK    128       // B*S
#define NCHUNK  256       // DIN / 32
#define CH_PER_WG 32
#define ROWH    40        // padded LDS row length in halfs (80 B)

// ---- branchless, loop-free decode: one index -> 2D point (2 weights) ----
// mod(a, m) for a>=0 computed as a - floor(a*inv_m)*m (same as jnp.mod).
// fmod(theta, U) is exact via th = fma(-floor(theta*invU), U, theta) since
// theta is an integer < 2^24 and U = 4096 (power of two).
// (b-c)/nf + c with nf = halfrl/dis  ==  (b-c)*(dis*inv_halfrl) + c.
__device__ __forceinline__ float2 decode_point(float theta, int Kv,
    float Uf, float invU, float ed, float rl, float inv_rl,
    float halfrl, float inv_halfrl,
    float ajx, float ajy, float nlx, float nly, float cx, float cy) {
  float th, scale;
  bool xform;
  if (Kv != 0) {                       // uniform branch
    float icl = floorf(theta * invU);  // index_class
    th    = fmaf(-icl, Uf, theta);     // theta mod U (exact)
    float dis = fmaf(icl, ed, halfrl);
    scale = dis * inv_halfrl;          // == 1/node_factor
    xform = (dis != halfrl);           // node_factor != 1
  } else {
    th = theta; scale = 1.0f; xform = false;
  }
  float ax = ajx * th;
  float ay = ajy * th;
  float bx = fmaf(-floorf(ax * inv_rl), rl, ax) + nlx;
  float by = fmaf(-floorf(ay * inv_rl), rl, ay) + nly;
  float tx = fmaf(bx - cx, scale, cx);
  float ty = fmaf(by - cy, scale, cy);
  bx = xform ? tx : bx;                // v_cndmask, no divergence
  by = xform ? ty : by;
  return make_float2(bx, by);
}

// ---- kernel 1: x (f32, 128x8192) -> f16 WMMA A-fragments in workspace ----
// A 16x32 f16 k order chosen here (matches B): lanes 0-15 hold M=0..15, half h -> K=16g+h
// (g = lane/16). One thread per (token-tile, k-chunk, lane).
__global__ __launch_bounds__(256) void hl_prep_a(const float* __restrict__ x,
                                                 _Float16* __restrict__ aFrag) {
  int tidg = blockIdx.x * 256 + threadIdx.x;        // 0..65535
  int lane = tidg & 31;
  int kb   = (tidg >> 5) & (NCHUNK - 1);            // 0..255
  int mt   = tidg >> 13;                            // 0..7 token tile
  int r = lane & 15, g = lane >> 4;
  const float* xrow = x + (size_t)(mt * 16 + r) * DIN + (size_t)kb * 32 + 16 * g;
  v16h v;
#pragma unroll
  for (int h = 0; h < 16; ++h) {
    int k = h;                                      // +16g folded into xrow
    v[h] = (_Float16)xrow[k];
  }
  *(v16h*)(aFrag + ((size_t)(mt * NCHUNK + kb) * 32 + lane) * 16) = v;
}

// ---- kernel 2: fused decode + GEMM + bias ----
// Grid: DOUT/32 = 256 workgroups, 256 threads (8 wave32). Wave w = token tile w.
// WG owns 32 channels; per 32-wide K chunk all threads decode a 32x32 f16 B tile
// into double-buffered LDS; each wave issues 2x v_wmma_f32_16x16x32_f16.
__global__ __launch_bounds__(256) void hl_gemm(
    const _Float16* __restrict__ aFrag,
    const int*   __restrict__ idxm,
    const float* __restrict__ each_dis, const float* __restrict__ rect_l,
    const float* __restrict__ node_ld,  const float* __restrict__ a_jump,
    const float* __restrict__ center,   const float* __restrict__ bias,
    const int*   __restrict__ Up,       const int* __restrict__ Kp,
    float* __restrict__ out) {
  __shared__ _Float16 ldsB[2][CH_PER_WG][ROWH];     // 2 x 32 x 80B = 5 KB

  const int tid    = threadIdx.x;
  const int lane   = tid & 31;
  const int wave   = tid >> 5;                      // token tile 0..7
  const int chBase = blockIdx.x * CH_PER_WG;
  const int r = lane & 15;
  const int g = lane >> 4;

  const float Uf   = (float)(*Up);
  const int   Kv   = *Kp;
  const float ed   = each_dis[0], rl = rect_l[0];
  const float ajx  = a_jump[0],  ajy = a_jump[1];
  const float nlx  = node_ld[0], nly = node_ld[1];
  const float cx   = center[0],  cy  = center[1];
  const float invU = 1.0f / Uf;            // U = 4096: exact
  const float inv_rl = 1.0f / rl;
  const float halfrl = 0.5f * rl;
  const float inv_halfrl = 2.0f * inv_rl;  // 1/(rl/2), exact scaling

  // decode assignment: thread handles 2 consecutive indices (4 weights)
  const int ich = tid >> 3;                 // local channel 0..31
  const int jin = (tid << 1) & 15;          // even j within chunk (16 idx/chunk)
  const long long irow = (long long)(chBase + ich) * K1 + jin;

  auto decodeChunk = [&](int kb, int buf) {
    int2 ii = *(const int2*)(idxm + irow + (long long)kb * 16);
    float2 p0 = decode_point((float)ii.x, Kv, Uf, invU, ed, rl, inv_rl,
                             halfrl, inv_halfrl, ajx, ajy, nlx, nly, cx, cy);
    float2 p1 = decode_point((float)ii.y, Kv, Uf, invU, ed, rl, inv_rl,
                             halfrl, inv_halfrl, ajx, ajy, nlx, nly, cx, cy);
    v4h w;
    w[0] = (_Float16)p0.x; w[1] = (_Float16)p0.y;
    w[2] = (_Float16)p1.x; w[3] = (_Float16)p1.y;
    *(v4h*)&ldsB[buf][ich][2 * jin] = w;            // 8B aligned ds_store_b64
  };

  v8f acc0 = {}; v8f acc1 = {};
  const _Float16* aBase = aFrag + (size_t)wave * NCHUNK * 512 + (size_t)lane * 16;

  decodeChunk(0, 0);
  __syncthreads();

  for (int kb = 0; kb < NCHUNK; ++kb) {
    int buf = kb & 1;
    if (kb + 2 < NCHUNK)
      __builtin_prefetch(idxm + irow + (long long)(kb + 2) * 16, 0, 1);
    if (kb + 1 < NCHUNK) decodeChunk(kb + 1, buf ^ 1);

    v16h a  = *(const v16h*)(aBase + (size_t)kb * 512);
    // B 32x16 f16 k order chosen here (matches A): N = lane%16, half h -> K = 16*(lane/16) + h
    v16h b0 = *(const v16h*)(&ldsB[buf][r][16 * g]);
    v16h b1 = *(const v16h*)(&ldsB[buf][16 + r][16 * g]);

    acc0 = __builtin_amdgcn_wmma_f32_16x16x32_f16(false, a, false, b0,
                                                  (short)0, acc0, false, false);
    acc1 = __builtin_amdgcn_wmma_f32_16x16x32_f16(false, a, false, b1,
                                                  (short)0, acc1, false, false);
    __syncthreads();
  }

  // C/D layout: VGPR i -> M = i + 8*g, lane%16 -> N
  float bv0 = bias[chBase + r];
  float bv1 = bias[chBase + 16 + r];
#pragma unroll
  for (int i = 0; i < 8; ++i) {
    int t = wave * 16 + i + 8 * g;
    out[(size_t)t * DOUT + chBase + r]      = acc0[i] + bv0;
    out[(size_t)t * DOUT + chBase + 16 + r] = acc1[i] + bv1;
  }
}

extern "C" void kernel_launch(void* const* d_in, const int* in_sizes, int n_in,
                              void* d_out, int out_size, void* d_ws, size_t ws_size,
                              hipStream_t stream) {
  const float* x        = (const float*)d_in[0];
  const int*   idxm     = (const int*)  d_in[1];
  const float* each_dis = (const float*)d_in[2];
  const float* rect_l   = (const float*)d_in[3];
  const float* node_ld  = (const float*)d_in[4];
  const float* a_jump   = (const float*)d_in[5];
  const float* center   = (const float*)d_in[6];
  const float* bias     = (const float*)d_in[7];
  const int*   Up       = (const int*)  d_in[8];
  const int*   Kp       = (const int*)  d_in[9];
  float*       out      = (float*)d_out;

  _Float16* aFrag = (_Float16*)d_ws;   // needs 8*256*512 halfs = 2 MB

  // 8 token tiles * 256 chunks * 32 lanes = 65536 threads
  hl_prep_a<<<256, 256, 0, stream>>>(x, aFrag);
  // 8192 channels / 32 per WG = 256 workgroups
  hl_gemm<<<256, 256, 0, stream>>>(aFrag, idxm, each_dis, rect_l, node_ld,
                                   a_jump, center, bias, Up, Kp, out);
}
